// RelativePositioningPitchEnergyEncoder_17205638987966
// MI455X (gfx1250) — matrix operand order repaired and checked by reading, not compiled, not run on hardware
//
#include <hip/hip_runtime.h>
#include <hip/hip_bf16.h>

#define B_   8
#define T_   768
#define C_   512
#define H_   8
#define FFN_ 2048
#define KCH_ 64
#define WIN_ 4
#define NEGV (-1e4f)

typedef __attribute__((ext_vector_type(16))) __bf16 v16bf;
typedef __attribute__((ext_vector_type(8)))  float  v8f;

// ---------------- bf16 helpers ----------------
__device__ __forceinline__ unsigned short f2bf(float f) {
  unsigned int u = __float_as_uint(f);
  u += 0x7FFFu + ((u >> 16) & 1u);  // round-to-nearest-even
  return (unsigned short)(u >> 16);
}
__device__ __forceinline__ float bf2f(unsigned short u) {
  return __uint_as_float(((unsigned int)u) << 16);
}

// ---------------- CDNA5 async global->LDS copy (ASYNCcnt) ----------------
__device__ __forceinline__ void cp16(unsigned ldsOff, const void* gsrc) {
  asm volatile("global_load_async_to_lds_b128 %0, %1, off"
               :: "v"(ldsOff), "v"(gsrc) : "memory");
}
__device__ __forceinline__ void wait_async() {
  asm volatile("s_wait_asynccnt 0x0" ::: "memory");
}

// ---------------- WMMA fragment loaders: 2x ds_load_b128 each ----------------
// A (16x32, MxK) in row-major LDS [m][k]: lane needs K runs {h8..h8+7} and {16+h8..+7}.
__device__ __forceinline__ v16bf load_frag_a(const unsigned short* lds, int mBase, int lda, int kBase) {
  const int lane = threadIdx.x & 31;
  const int half = lane >> 4;
  const int row  = lane & 15;
  const unsigned short* p = lds + (mBase + row) * lda + kBase + half * 8;
  union { uint4 q[2]; v16bf v; } u;
  u.q[0] = *(const uint4*)p;          // K = kBase+half*8 .. +7
  u.q[1] = *(const uint4*)(p + 16);   // K = kBase+16+half*8 .. +7
  return u.v;
}
// B (32x16, KxN) in LDS stored [n][k]: lane needs K runs {h8..h8+7} and {16+h8..+7} of row n.
__device__ __forceinline__ v16bf load_frag_b(const unsigned short* lds, int kBase, int ldb, int nBase) {
  const int lane  = threadIdx.x & 31;
  const int khalf = lane >> 4;
  const int n     = lane & 15;
  const unsigned short* p = lds + (nBase + n) * ldb + kBase + khalf * 8;
  union { uint4 q[2]; v16bf v; } u;
  u.q[0] = *(const uint4*)p;
  u.q[1] = *(const uint4*)(p + 16);
  return u.v;
}

// ---------------- prep / convert kernels ----------------
__global__ __launch_bounds__(256) void mask_kernel(float* __restrict__ m, const int* __restrict__ lens) {
  int idx = blockIdx.x * 256 + threadIdx.x;
  if (idx >= B_ * T_) return;
  int b = idx / T_, t = idx % T_;
  m[idx] = (t < lens[b]) ? 1.0f : 0.0f;
}
__global__ __launch_bounds__(256) void prep_kernel(const float* __restrict__ x, const float* __restrict__ m,
                                                   float* __restrict__ xt) {
  long idx = (long)blockIdx.x * 256 + threadIdx.x;
  if (idx >= (long)B_ * C_ * T_) return;
  int b = (int)(idx / ((long)C_ * T_));
  int rem = (int)(idx % ((long)C_ * T_));
  int c = rem / T_, t = rem % T_;
  xt[idx] = x[((long)b * T_ + t) * C_ + c] * m[b * T_ + t];
}
__global__ __launch_bounds__(256) void mask_mul_kernel(float* __restrict__ xio, const float* __restrict__ m) {
  long idx = (long)blockIdx.x * 256 + threadIdx.x;
  if (idx >= (long)B_ * C_ * T_) return;
  int b = (int)(idx / ((long)C_ * T_));
  int t = (int)(idx % T_);
  xio[idx] *= m[b * T_ + t];
}
// xt [b][c][t] fp32 -> dst [b][t][c] bf16, masked
__global__ __launch_bounds__(256) void cvt_actT_kernel(unsigned short* __restrict__ dst,
                                                       const float* __restrict__ x, const float* __restrict__ m) {
  long idx = (long)blockIdx.x * 256 + threadIdx.x;
  if (idx >= (long)B_ * T_ * C_) return;
  int c = (int)(idx % C_);
  long bt = idx / C_;
  int t = (int)(bt % T_), b = (int)(bt / T_);
  dst[idx] = f2bf(x[((long)b * C_ + c) * T_ + t] * m[b * T_ + t]);
}
__global__ __launch_bounds__(256) void cvt_plain_kernel(unsigned short* __restrict__ dst,
                                                        const float* __restrict__ src, long n) {
  long idx = (long)blockIdx.x * 256 + threadIdx.x;
  if (idx < n) dst[idx] = f2bf(src[idx]);
}
// w [M][K][3] fp32 -> dst [tap][M][K] bf16
__global__ __launch_bounds__(256) void cvt_convw_kernel(unsigned short* __restrict__ dst,
                                                        const float* __restrict__ src, int M, int K) {
  long idx = (long)blockIdx.x * 256 + threadIdx.x;
  long mk = (long)M * K;
  if (idx >= 3 * mk) return;
  int tap = (int)(idx / mk);
  long rem = idx % mk;
  dst[idx] = f2bf(src[rem * 3 + tap]);
}

// ---------------- GEMM: Y[b] = W[MxK](bf16) * Xt[b][N rows][K](bf16) + bias ----------------
// OUTMODE 0: fp32 [M][N]; 1: bf16 [M][N]; 2: bf16 transposed [N][M]
template <int OUTMODE>
__global__ __launch_bounds__(256) void gemm_bf16_kernel(const unsigned short* __restrict__ A,
                                                        const unsigned short* __restrict__ Bact,
                                                        const float* __restrict__ bias, void* __restrict__ Y,
                                                        int M, int N, int K) {
  __shared__ __align__(16) unsigned short lA[64 * 40];
  __shared__ __align__(16) unsigned short lB[32 * 40];
  const int bz = blockIdx.z;
  const unsigned short* Bb = Bact + (long)bz * N * K;
  const int mBase = blockIdx.x * 64;
  const int nBase = blockIdx.y * 32;
  const int tid = threadIdx.x;
  const int wid = tid >> 5, wm = wid & 3, wn = wid >> 2;
  const unsigned laBase = (unsigned)(size_t)(&lA[0]);
  const unsigned lbBase = (unsigned)(size_t)(&lB[0]);
  v8f acc = {};
  for (int k0 = 0; k0 < K; k0 += 32) {
    { int r = tid >> 2, cch = tid & 3;            // 64x32 = 256 chunks of 16B
      int gm = mBase + r; if (gm >= M) gm = M - 1;
      cp16(laBase + (unsigned)((r * 40 + cch * 8) * 2), A + (long)gm * K + k0 + cch * 8); }
    if (tid < 128) {                              // 32x32 = 128 chunks
      int r = tid >> 2, cch = tid & 3;
      cp16(lbBase + (unsigned)((r * 40 + cch * 8) * 2), Bb + (long)(nBase + r) * K + k0 + cch * 8); }
    wait_async();
    __syncthreads();
    v16bf fa = load_frag_a(lA, wm * 16, 40, 0);
    v16bf fb = load_frag_b(lB, 0, 40, wn * 16);
    acc = __builtin_amdgcn_wmma_f32_16x16x32_bf16(false, fa, false, fb, (short)0, acc, false, false);
    __syncthreads();
  }
  const int lane = tid & 31, n = lane & 15, mh = lane >> 4;
#pragma unroll
  for (int r = 0; r < 8; ++r) {
    int row = mBase + wm * 16 + r + mh * 8;
    int col = nBase + wn * 16 + n;
    if (row >= M) continue;
    float v = acc[r] + bias[row];
    if (OUTMODE == 0) ((float*)Y)[(long)bz * M * N + (long)row * N + col] = v;
    if (OUTMODE == 1) ((unsigned short*)Y)[(long)bz * M * N + (long)row * N + col] = f2bf(v);
    if (OUTMODE == 2) ((unsigned short*)Y)[((long)bz * N + col) * M + row] = f2bf(v);
  }
}

// ---------------- KS=3 conv: Wt [tap][M][K] bf16, act [b][t][K] bf16 (pre-masked) ----------------
// OUTT 1: bf16 transposed [t][M] (masked, for FFN hidden); OUTT 0: fp32 [M][T] masked.
template <int RELU, int OUTT>
__global__ __launch_bounds__(256) void conv3_bf16_kernel(const unsigned short* __restrict__ Wt,
                                                         const unsigned short* __restrict__ act,
                                                         const float* __restrict__ bias,
                                                         const float* __restrict__ mask, void* __restrict__ Y,
                                                         int M, int K) {
  __shared__ __align__(16) unsigned short lA[3 * 64 * 40];
  __shared__ __align__(16) unsigned short lB[34 * 40];
  const int bz = blockIdx.z;
  const unsigned short* Bb = act + (long)bz * T_ * K;
  const float* msk = mask + (long)bz * T_;
  const int mBase = blockIdx.x * 64;
  const int nBase = blockIdx.y * 32;
  const int tid = threadIdx.x;
  const int wid = tid >> 5, wm = wid & 3, wn = wid >> 2;
  const unsigned laBase = (unsigned)(size_t)(&lA[0]);
  const unsigned lbBase = (unsigned)(size_t)(&lB[0]);
  // zero halo rows (t = -1 or t = T) once; async copies skip them below
  if (tid < 64) {
    int n = (tid >> 5) ? 33 : 0;
    int t = nBase - 1 + n;
    if (t < 0 || t >= T_) lB[n * 40 + (tid & 31)] = 0;
  }
  __syncthreads();
  v8f acc = {};
  for (int k0 = 0; k0 < K; k0 += 32) {
#pragma unroll
    for (int j = 0; j < 3; ++j) {                 // 3 x 256 A chunks
      int i = tid + 256 * j;
      int tap = i >> 8, rem = i & 255;
      int r = rem >> 2, cch = rem & 3;
      int gm = mBase + r; if (gm >= M) gm = M - 1;
      cp16(laBase + (unsigned)(((tap * 64 + r) * 40 + cch * 8) * 2),
           Wt + ((long)tap * M + gm) * K + k0 + cch * 8);
    }
    if (tid < 136) {                              // 34 rows x 4 chunks
      int r = tid >> 2, cch = tid & 3;
      int t = nBase - 1 + r;
      if (t >= 0 && t < T_)
        cp16(lbBase + (unsigned)((r * 40 + cch * 8) * 2), Bb + (long)t * K + k0 + cch * 8);
    }
    wait_async();
    __syncthreads();
#pragma unroll
    for (int tap = 0; tap < 3; ++tap) {
      v16bf fa = load_frag_a(lA + tap * 64 * 40, wm * 16, 40, 0);
      v16bf fb = load_frag_b(lB, 0, 40, wn * 16 + tap);
      acc = __builtin_amdgcn_wmma_f32_16x16x32_bf16(false, fa, false, fb, (short)0, acc, false, false);
    }
    __syncthreads();
  }
  const int lane = tid & 31, n = lane & 15, mh = lane >> 4;
#pragma unroll
  for (int r = 0; r < 8; ++r) {
    int row = mBase + wm * 16 + r + mh * 8;
    int col = nBase + wn * 16 + n;
    if (row >= M) continue;
    float v = acc[r] + bias[row];
    if (RELU) v = fmaxf(v, 0.0f);
    v *= msk[col];
    if (OUTT) ((unsigned short*)Y)[((long)bz * T_ + col) * M + row] = f2bf(v);
    else      ((float*)Y)[(long)bz * M * T_ + (long)row * T_ + col] = v;
  }
}

// ---------------- attention scores: S[bh][t,s] = scale * qT[t]·kT[s] over d=64 ----------------
__global__ __launch_bounds__(256) void attn_score_kernel(const unsigned short* __restrict__ qT,
                                                         const unsigned short* __restrict__ kT,
                                                         float* __restrict__ S, float scale) {
  __shared__ __align__(16) unsigned short lA[64 * 72];  // [t][d]
  __shared__ __align__(16) unsigned short lB[32 * 72];  // [s][d]
  const int bh = blockIdx.z, b = bh >> 3, h = bh & 7;
  const long bOff = (long)b * T_;
  const int mBase = blockIdx.x * 64;
  const int nBase = blockIdx.y * 32;
  const int tid = threadIdx.x;
  const int wid = tid >> 5, wm = wid & 3, wn = wid >> 2;
  const unsigned laBase = (unsigned)(size_t)(&lA[0]);
  const unsigned lbBase = (unsigned)(size_t)(&lB[0]);
#pragma unroll
  for (int j = 0; j < 2; ++j) {                   // 64 rows x 8 chunks
    int i = tid + 256 * j;
    int r = i >> 3, cch = i & 7;
    cp16(laBase + (unsigned)((r * 72 + cch * 8) * 2),
         qT + (bOff + mBase + r) * C_ + h * KCH_ + cch * 8);
  }
  { int r = tid >> 3, cch = tid & 7;              // 32 rows x 8 chunks
    cp16(lbBase + (unsigned)((r * 72 + cch * 8) * 2),
         kT + (bOff + nBase + r) * C_ + h * KCH_ + cch * 8); }
  wait_async();
  __syncthreads();
  v8f acc = {};
#pragma unroll
  for (int ks = 0; ks < 64; ks += 32) {
    v16bf fa = load_frag_a(lA, wm * 16, 72, ks);
    v16bf fb = load_frag_b(lB, ks, 72, wn * 16);
    acc = __builtin_amdgcn_wmma_f32_16x16x32_bf16(false, fa, false, fb, (short)0, acc, false, false);
  }
  const int lane = tid & 31, n = lane & 15, mh = lane >> 4;
  float* Sb = S + (long)bh * T_ * T_;
#pragma unroll
  for (int r = 0; r < 8; ++r) {
    int row = mBase + wm * 16 + r + mh * 8;
    int col = nBase + wn * 16 + n;
    Sb[(long)row * T_ + col] = acc[r] * scale;
  }
}

// ---------------- banded rel-K scores ----------------
__global__ __launch_bounds__(256) void rel_score_kernel(float* __restrict__ S, const unsigned short* __restrict__ qT,
                                                        const float* __restrict__ embk, float scale) {
  long idx = (long)blockIdx.x * 256 + threadIdx.x;
  const long total = (long)B_ * H_ * T_ * (2 * WIN_ + 1);
  if (idx >= total) return;
  int dlt = (int)(idx % (2 * WIN_ + 1));
  long r2 = idx / (2 * WIN_ + 1);
  int t = (int)(r2 % T_);
  int bh = (int)(r2 / T_), b = bh >> 3, h = bh & 7;
  int s = t + dlt - WIN_;
  if (s < 0 || s >= T_) return;
  const unsigned short* q = qT + ((long)b * T_ + t) * C_ + h * KCH_;
  float acc = 0.0f;
#pragma unroll 8
  for (int d = 0; d < KCH_; ++d) acc += bf2f(q[d]) * embk[dlt * KCH_ + d];
  S[((long)bh * T_ + t) * T_ + s] += acc * scale;
}

// ---------------- masked softmax: fp32 in, bf16 probabilities out ----------------
__global__ __launch_bounds__(256) void softmax_kernel(const float* __restrict__ S, unsigned short* __restrict__ P,
                                                      const float* __restrict__ mask) {
  const int wid = threadIdx.x >> 5, lane = threadIdx.x & 31;
  const long row = (long)blockIdx.x * 8 + wid;
  const int t = (int)(row % T_);
  const int bh = (int)(row / T_), b = bh >> 3;
  const float mt = mask[b * T_ + t];
  const float* Srow = S + row * T_;
  unsigned short* Prow = P + row * T_;
  float vals[T_ / 32];
  float mx = -3.4e38f;
#pragma unroll
  for (int i = 0; i < T_ / 32; ++i) {
    int s = lane + i * 32;
    float ms = mask[b * T_ + s];
    float v = (mt * ms == 0.0f) ? NEGV : Srow[s];
    vals[i] = v;
    mx = fmaxf(mx, v);
  }
#pragma unroll
  for (int o = 16; o >= 1; o >>= 1) mx = fmaxf(mx, __shfl_xor(mx, o, 32));
  float sum = 0.0f;
#pragma unroll
  for (int i = 0; i < T_ / 32; ++i) { float e = __expf(vals[i] - mx); vals[i] = e; sum += e; }
#pragma unroll
  for (int o = 16; o >= 1; o >>= 1) sum += __shfl_xor(sum, o, 32);
  const float inv = 1.0f / sum;
#pragma unroll
  for (int i = 0; i < T_ / 32; ++i) Prow[lane + i * 32] = f2bf(vals[i] * inv);
}

// ---------------- attention output: att[t][c] (fp32, transposed) = P x V ----------------
__global__ __launch_bounds__(256) void attn_out_kernel(const unsigned short* __restrict__ P,
                                                       const unsigned short* __restrict__ V,
                                                       float* __restrict__ att) {
  __shared__ __align__(16) unsigned short lA[64 * 40];  // [t][s]
  __shared__ __align__(16) unsigned short lB[32 * 40];  // [d][s]
  const int bh = blockIdx.z, b = bh >> 3, h = bh & 7;
  const unsigned short* Pb = P + (long)bh * T_ * T_;
  const unsigned short* Vb = V + ((long)b * C_ + h * KCH_) * T_;
  const int mBase = blockIdx.x * 64;  // t
  const int nBase = blockIdx.y * 32;  // d
  const int tid = threadIdx.x;
  const int wid = tid >> 5, wm = wid & 3, wn = wid >> 2;
  const unsigned laBase = (unsigned)(size_t)(&lA[0]);
  const unsigned lbBase = (unsigned)(size_t)(&lB[0]);
  v8f acc = {};
  for (int k0 = 0; k0 < T_; k0 += 32) {
    { int r = tid >> 2, cch = tid & 3;            // 64 rows x 4 chunks
      cp16(laBase + (unsigned)((r * 40 + cch * 8) * 2), Pb + (long)(mBase + r) * T_ + k0 + cch * 8); }
    if (tid < 128) {                              // 32 rows x 4 chunks
      int r = tid >> 2, cch = tid & 3;
      cp16(lbBase + (unsigned)((r * 40 + cch * 8) * 2), Vb + (long)(nBase + r) * T_ + k0 + cch * 8); }
    wait_async();
    __syncthreads();
    v16bf fa = load_frag_a(lA, wm * 16, 40, 0);
    v16bf fb = load_frag_b(lB, 0, 40, wn * 16);
    acc = __builtin_amdgcn_wmma_f32_16x16x32_bf16(false, fa, false, fb, (short)0, acc, false, false);
    __syncthreads();
  }
  const int lane = tid & 31, n = lane & 15, mh = lane >> 4;
#pragma unroll
  for (int r = 0; r < 8; ++r) {
    int trow = mBase + wm * 16 + r + mh * 8;
    int dcol = nBase + wn * 16 + n;
    att[((long)b * T_ + trow) * C_ + h * KCH_ + dcol] = acc[r];
  }
}

// ---------------- banded rel-V output (att layout [t][c]) ----------------
__global__ __launch_bounds__(256) void rel_out_kernel(float* __restrict__ att, const unsigned short* __restrict__ P,
                                                      const float* __restrict__ embv) {
  long idx = (long)blockIdx.x * 256 + threadIdx.x;
  const long total = (long)B_ * H_ * T_ * KCH_;
  if (idx >= total) return;
  int d = (int)(idx & (KCH_ - 1));
  long r2 = idx >> 6;
  int t = (int)(r2 % T_);
  int bh = (int)(r2 / T_), b = bh >> 3, h = bh & 7;
  const unsigned short* Prow = P + ((long)bh * T_ + t) * T_;
  float acc = 0.0f;
#pragma unroll
  for (int dlt = 0; dlt < 2 * WIN_ + 1; ++dlt) {
    int s = t + dlt - WIN_;
    if (s >= 0 && s < T_) acc += bf2f(Prow[s]) * embv[dlt * KCH_ + d];
  }
  att[((long)b * T_ + t) * C_ + h * KCH_ + d] += acc;
}

// ---------------- residual + LayerNorm over channels (fp32 [c][t]) ----------------
__global__ __launch_bounds__(256) void add_ln_kernel(float* __restrict__ Xio, const float* __restrict__ Yin,
                                                     const float* __restrict__ g, const float* __restrict__ bta) {
  int idx = blockIdx.x * 256 + threadIdx.x;
  if (idx >= B_ * T_) return;
  int b = idx / T_, t = idx % T_;
  float* xb = Xio + (long)b * C_ * T_ + t;
  const float* yb = Yin + (long)b * C_ * T_ + t;
  float s = 0.0f, ss = 0.0f;
  for (int c = 0; c < C_; ++c) {
    float v = xb[(long)c * T_] + yb[(long)c * T_];
    s += v; ss += v * v;
  }
  const float mu = s * (1.0f / C_);
  const float var = ss * (1.0f / C_) - mu * mu;
  const float inv = rsqrtf(var + 1e-5f);
  for (int c = 0; c < C_; ++c) {
    float v = xb[(long)c * T_] + yb[(long)c * T_];
    xb[(long)c * T_] = (v - mu) * inv * g[c] + bta[c];
  }
}

// ---------------- exact final: singleton-channel LN collapses to b2 ----------------
__global__ __launch_bounds__(256) void final_kernel(float* __restrict__ out, const int* __restrict__ lens,
                                                    const float* __restrict__ b2) {
  int idx = blockIdx.x * 256 + threadIdx.x;
  if (idx >= B_ * T_) return;
  int b = idx / T_, t = idx % T_;
  out[idx] = (t < lens[b]) ? b2[0] : 0.0f;
}

// ---------------- host launch ----------------
extern "C" void kernel_launch(void* const* d_in, const int* in_sizes, int n_in,
                              void* d_out, int out_size, void* d_ws, size_t ws_size,
                              hipStream_t stream) {
  (void)in_sizes; (void)n_in; (void)out_size;
  const float* x  = (const float*)d_in[0];
  const int* lens = (const int*)d_in[1];
  // jax pytree flatten -> sorted dict keys per layer (18 leaves; 20 in last layer):
  // b1 b2 b_f1 b_f2 [b_proj] bk bo bq bv emb_k emb_v g1 g2 w_f1 w_f2 [w_proj] wk wo wq wv
  auto P = [&](int L, int off) -> const float* { return (const float*)d_in[2 + L * 18 + off]; };
  const float* b2_last = (const float*)d_in[2 + 5 * 18 + 1];

  const long fBT  = (long)B_ * T_;
  const long fBCT = (long)B_ * C_ * T_;
  const long fS   = (long)B_ * H_ * T_ * T_;
  const long fH   = (long)B_ * FFN_ * T_;

  // workspace layout (64B aligned slots)
  char* base = (char*)d_ws;
  long off = 0;
  auto alloc = [&](long bytes) -> char* { char* p = base + off; off += (bytes + 63) & ~63L; return p; };
  float* m    = (float*)alloc(fBT * 4);
  float* xt   = (float*)alloc(fBCT * 4);
  float* att  = (float*)alloc(fBCT * 4);   // [b][t][c]
  float* yb   = (float*)alloc(fBCT * 4);   // [b][c][t]
  float* sc   = (float*)alloc(fS * 4);
  unsigned short* xtb  = (unsigned short*)alloc(fBCT * 2);  // [b][t][c] masked
  unsigned short* qT   = (unsigned short*)alloc(fBCT * 2);  // [b][t][c]
  unsigned short* kT   = (unsigned short*)alloc(fBCT * 2);  // [b][t][c]
  unsigned short* vbb  = (unsigned short*)alloc(fBCT * 2);  // [b][c][t]
  unsigned short* attb = (unsigned short*)alloc(fBCT * 2);  // [b][t][c]
  unsigned short* hbT  = (unsigned short*)alloc(fH * 2);    // [b][t][FFN]
  unsigned short* scb  = (unsigned short*)alloc(fS * 2);    // probabilities
  unsigned short* wqb  = (unsigned short*)alloc((long)C_ * C_ * 2);
  unsigned short* wkb  = (unsigned short*)alloc((long)C_ * C_ * 2);
  unsigned short* wvb  = (unsigned short*)alloc((long)C_ * C_ * 2);
  unsigned short* wob  = (unsigned short*)alloc((long)C_ * C_ * 2);
  unsigned short* wf1b = (unsigned short*)alloc((long)3 * FFN_ * C_ * 2);
  unsigned short* wf2b = (unsigned short*)alloc((long)3 * C_ * FFN_ * 2);
  const long need = off;

  const int nBT = (int)((fBT + 255) / 256);
  if ((long)ws_size < need) {  // not enough scratch: output is closed-form anyway
    final_kernel<<<nBT, 256, 0, stream>>>((float*)d_out, lens, b2_last);
    return;
  }

  const int nBCT = (int)((fBCT + 255) / 256);
  const int nCC  = (int)(((long)C_ * C_ + 255) / 256);
  const float scale = 0.125f;  // KCH^-0.5

  mask_kernel<<<nBT, 256, 0, stream>>>(m, lens);
  prep_kernel<<<nBCT, 256, 0, stream>>>(x, m, xt);

  for (int L = 0; L < 5; ++L) {
    mask_mul_kernel<<<nBCT, 256, 0, stream>>>(xt, m);
    cvt_actT_kernel<<<nBCT, 256, 0, stream>>>(xtb, xt, m);
    cvt_plain_kernel<<<nCC, 256, 0, stream>>>(wqb, P(L, 16), (long)C_ * C_);
    cvt_plain_kernel<<<nCC, 256, 0, stream>>>(wkb, P(L, 14), (long)C_ * C_);
    cvt_plain_kernel<<<nCC, 256, 0, stream>>>(wvb, P(L, 17), (long)C_ * C_);
    cvt_plain_kernel<<<nCC, 256, 0, stream>>>(wob, P(L, 15), (long)C_ * C_);
    // Q,K transposed bf16; V normal bf16
    gemm_bf16_kernel<2><<<dim3(C_ / 64, T_ / 32, B_), 256, 0, stream>>>(wqb, xtb, P(L, 6), qT, C_, T_, C_);
    gemm_bf16_kernel<2><<<dim3(C_ / 64, T_ / 32, B_), 256, 0, stream>>>(wkb, xtb, P(L, 4), kT, C_, T_, C_);
    gemm_bf16_kernel<1><<<dim3(C_ / 64, T_ / 32, B_), 256, 0, stream>>>(wvb, xtb, P(L, 7), vbb, C_, T_, C_);
    // scores + banded rel-K + softmax(->bf16 P)
    attn_score_kernel<<<dim3(T_ / 64, T_ / 32, B_ * H_), 256, 0, stream>>>(qT, kT, sc, scale);
    rel_score_kernel<<<(int)(((long)B_ * H_ * T_ * 9 + 255) / 256), 256, 0, stream>>>(sc, qT, P(L, 8), scale);
    softmax_kernel<<<B_ * H_ * T_ / 8, 256, 0, stream>>>(sc, scb, m);
    // P x V + banded rel-V, then convert att -> bf16
    attn_out_kernel<<<dim3(T_ / 64, KCH_ / 32, B_ * H_), 256, 0, stream>>>(scb, vbb, att);
    rel_out_kernel<<<(int)(((long)B_ * H_ * T_ * KCH_ + 255) / 256), 256, 0, stream>>>(att, scb, P(L, 9));
    cvt_plain_kernel<<<nBCT, 256, 0, stream>>>(attb, att, fBCT);
    // output projection (fp32), residual LN1
    gemm_bf16_kernel<0><<<dim3(C_ / 64, T_ / 32, B_), 256, 0, stream>>>(wob, attb, P(L, 5), yb, C_, T_, C_);
    add_ln_kernel<<<nBT, 256, 0, stream>>>(xt, yb, P(L, 10), P(L, 0));
    // FFN convs on masked post-LN1 activations
    cvt_actT_kernel<<<nBCT, 256, 0, stream>>>(xtb, xt, m);
    cvt_convw_kernel<<<(int)(((long)3 * FFN_ * C_ + 255) / 256), 256, 0, stream>>>(wf1b, P(L, 12), FFN_, C_);
    cvt_convw_kernel<<<(int)(((long)3 * C_ * FFN_ + 255) / 256), 256, 0, stream>>>(wf2b, P(L, 13), C_, FFN_);
    conv3_bf16_kernel<1, 1><<<dim3(FFN_ / 64, T_ / 32, B_), 256, 0, stream>>>(wf1b, xtb, P(L, 2), m, hbT, FFN_, C_);
    conv3_bf16_kernel<0, 0><<<dim3(C_ / 64, T_ / 32, B_), 256, 0, stream>>>(wf2b, hbT, P(L, 3), m, yb, C_, FFN_);
    add_ln_kernel<<<nBT, 256, 0, stream>>>(xt, yb, P(L, 11), P(L, 1));
  }
  // Layer 5 projects to 1 channel; LN over a singleton channel yields exactly b2 broadcast.
  final_kernel<<<nBT, 256, 0, stream>>>((float*)d_out, lens, b2_last);
}
